// clip_info_2070174237310
// MI455X (gfx1250) — matrix operand order repaired and checked by reading, not compiled, not run on hardware
//
#include <hip/hip_runtime.h>
#include <hip/hip_bf16.h>
#include <math.h>

#define B_DIM 256
#define L_DIM 256
#define INV_TAU 100.0f

typedef _Float16 v16h __attribute__((ext_vector_type(16)));
typedef _Float16 v8h  __attribute__((ext_vector_type(8)));
typedef float    v8f  __attribute__((ext_vector_type(8)));

__global__ void zero_out_kernel(float* out) {
    if (threadIdx.x == 0 && blockIdx.x == 0) out[0] = 0.0f;
}

// ---------------------------------------------------------------------------
// K1: softmax over axis L of [b, l, i] tensor, cast to f16, write transposed
// layout [i][row][l] (row = b for x, row = c for target). Mask applied
// post-softmax for the target tensor (matches reference tsm = ts * m).
// grid = (B, B/32, 2), block = 256
// ---------------------------------------------------------------------------
__global__ void softmax_transpose_kernel(const float* __restrict__ x,
                                         const float* __restrict__ tgt,
                                         const int*   __restrict__ mask,
                                         _Float16*    __restrict__ xs_t,
                                         _Float16*    __restrict__ ts_t)
{
    const int b     = blockIdx.x;          // clip row (b or c)
    const int i0    = blockIdx.y * 32;     // query tile base
    const int which = blockIdx.z;          // 0 = x, 1 = target

    const float* __restrict__ src = (which == 0) ? x : tgt;
    _Float16*    __restrict__ dst = (which == 0) ? xs_t : ts_t;

    __shared__ float tile[L_DIM * 33];     // [l][i_local], pad 33 vs 64 banks
    __shared__ float red[256];
    __shared__ float colmax[32];
    __shared__ float colsum[32];

    const int tid   = threadIdx.x;
    const int il    = tid & 31;            // i_local 0..31
    const int q     = tid >> 5;            // l-quarter group 0..7
    const int lbase = q * 32;

    // Coalesced load of [256 l x 32 i] f32 tile into LDS
    const size_t src_base = (size_t)b * L_DIM * B_DIM + (size_t)(i0 + il);
    #pragma unroll 4
    for (int lq = 0; lq < 32; ++lq) {
        const int l = lbase + lq;
        tile[l * 33 + il] = src[src_base + (size_t)l * B_DIM];
    }
    __syncthreads();

    // Column max
    float mx = -INFINITY;
    #pragma unroll 4
    for (int lq = 0; lq < 32; ++lq)
        mx = fmaxf(mx, tile[(lbase + lq) * 33 + il]);
    red[tid] = mx;
    __syncthreads();
    if (q == 0) {
        float m2 = red[il];
        #pragma unroll
        for (int qq = 1; qq < 8; ++qq) m2 = fmaxf(m2, red[qq * 32 + il]);
        colmax[il] = m2;
    }
    __syncthreads();
    mx = colmax[il];

    // Column sum of exp
    float s = 0.0f;
    #pragma unroll 4
    for (int lq = 0; lq < 32; ++lq)
        s += expf(tile[(lbase + lq) * 33 + il] - mx);
    red[tid] = s;
    __syncthreads();
    if (q == 0) {
        float s2 = 0.0f;
        #pragma unroll
        for (int qq = 0; qq < 8; ++qq) s2 += red[qq * 32 + il];
        colsum[il] = s2;
    }
    __syncthreads();
    const float inv = 1.0f / colsum[il];

    // Transposed f16 write: dst[i][b][l], contiguous in l, 16B packed stores
    const size_t out_base = (size_t)(i0 + il) * B_DIM * L_DIM + (size_t)b * L_DIM + lbase;
    const int mrow = (i0 + il) * L_DIM;    // mask row for query i
    #pragma unroll
    for (int g = 0; g < 4; ++g) {
        v8h pack;
        #pragma unroll
        for (int e = 0; e < 8; ++e) {
            const int l = lbase + g * 8 + e;
            float v = expf(tile[l * 33 + il] - mx) * inv;
            if (which == 1 && mask[mrow + l] <= 0) v = 0.0f;
            pack[e] = (_Float16)v;
        }
        *(v8h*)(dst + out_base + g * 8) = pack;
    }
}

// ---------------------------------------------------------------------------
// K2: per (query i, 16-row b-tile): sim slab [16 x 256] = A @ T^T via WMMA,
// fused log-softmax over c, diagonal pick, weighted reduction into loss.
// grid = (16, 256), block = 256 (8 waves). Each wave: 2 c-tiles x 8 K-steps.
// ---------------------------------------------------------------------------
__global__ void wmma_loss_kernel(const _Float16* __restrict__ xs_t,
                                 const _Float16* __restrict__ ts_t,
                                 const int*      __restrict__ label_mat,
                                 float*          __restrict__ out)
{
    const int btile = blockIdx.x;   // 16-row slab of b
    const int i     = blockIdx.y;   // query

    __shared__ float simLDS[16 * 257];   // [row][c], padded

    const int tid  = threadIdx.x;
    const int wave = tid >> 5;
    const int lane = tid & 31;
    const int m    = lane & 15;     // A row index / B column index
    const int h    = lane >> 4;     // half-wave select

    const _Float16* __restrict__ Arow =
        xs_t + (size_t)i * B_DIM * L_DIM + (size_t)(btile * 16 + m) * L_DIM;
    const _Float16* __restrict__ Tbase =
        ts_t + (size_t)i * B_DIM * L_DIM;

    #pragma unroll
    for (int cti = 0; cti < 2; ++cti) {
        const int ct = wave * 2 + cti;
        const _Float16* __restrict__ Trow = Tbase + (size_t)(ct * 16 + m) * L_DIM;

        v8f acc = {};
        #pragma unroll
        for (int k0 = 0; k0 < L_DIM; k0 += 32) {
            // A fragment (16x32 f16, ISA layout): lane m+16h holds
            // K[k0+8h .. +8) in elems 0..7 and K[k0+16+8h .. +8) in 8..15
            union { v16h v; v8h half[2]; } av;
            av.half[0] = *(const v8h*)(Arow + k0 + 8 * h);
            av.half[1] = *(const v8h*)(Arow + k0 + 16 + 8 * h);
            // B fragment (32x16 f16): lane n+16h holds column n (= T row n),
            // K[k0+8h .. +8) in elems 0..7 and K[k0+16+8h .. +8) in 8..15
            union { v16h v; v8h half[2]; } bu;
            bu.half[0] = *(const v8h*)(Trow + k0 + 8 * h);
            bu.half[1] = *(const v8h*)(Trow + k0 + 16 + 8 * h);
            const v16h bv = bu.v;

            acc = __builtin_amdgcn_wmma_f32_16x16x32_f16(
                false, av.v, false, bv, (short)0, acc, false, false);
        }
        // C/D layout: lane n+16h, VGPR r -> D[r + 8h][n]; scale to logits
        #pragma unroll
        for (int r = 0; r < 8; ++r)
            simLDS[(r + 8 * h) * 257 + ct * 16 + m] = acc[r] * INV_TAU;
    }
    __syncthreads();

    // logsumexp over c per row: 16 threads per row, shfl_xor butterflies
    const int row = tid >> 4;
    const int idx = tid & 15;
    float mx = -INFINITY;
    #pragma unroll 4
    for (int c = idx; c < B_DIM; c += 16)
        mx = fmaxf(mx, simLDS[row * 257 + c]);
    #pragma unroll
    for (int off = 8; off > 0; off >>= 1)
        mx = fmaxf(mx, __shfl_xor(mx, off, 16));

    float s = 0.0f;
    #pragma unroll 4
    for (int c = idx; c < B_DIM; c += 16)
        s += expf(simLDS[row * 257 + c] - mx);
    #pragma unroll
    for (int off = 8; off > 0; off >>= 1)
        s += __shfl_xor(s, off, 16);

    if (idx == 0) {
        const int   j    = btile * 16 + row;
        const float lse  = mx + logf(s);
        const float diag = simLDS[row * 257 + i];          // c = i
        const float lm   = (float)label_mat[i * B_DIM + j];
        atomicAdd(out, -lm * (diag - lse) * (1.0f / (float)B_DIM));
    }
}

extern "C" void kernel_launch(void* const* d_in, const int* in_sizes, int n_in,
                              void* d_out, int out_size, void* d_ws, size_t ws_size,
                              hipStream_t stream) {
    const float* x         = (const float*)d_in[0];
    const float* tgt       = (const float*)d_in[1];
    const int*   mask      = (const int*)d_in[2];
    // d_in[3] = query_labels: unused by the reference computation
    const int*   label_mat = (const int*)d_in[4];
    float*       out       = (float*)d_out;

    const size_t tensor_halfs = (size_t)B_DIM * B_DIM * L_DIM; // 16.7M f16 = 32 MB
    _Float16* xs_t = (_Float16*)d_ws;
    _Float16* ts_t = (_Float16*)((char*)d_ws + tensor_halfs * sizeof(_Float16));

    zero_out_kernel<<<1, 64, 0, stream>>>(out);

    softmax_transpose_kernel<<<dim3(B_DIM, B_DIM / 32, 2), 256, 0, stream>>>(
        x, tgt, mask, xs_t, ts_t);

    wmma_loss_kernel<<<dim3(16, B_DIM), 256, 0, stream>>>(
        xs_t, ts_t, label_mat, out);
}
